// SnailAttentionBlock_81484119540097
// MI455X (gfx1250) — matrix operand order repaired and checked by reading, not compiled, not run on hardware
//
#include <hip/hip_runtime.h>
#include <hip/hip_bf16.h>

typedef __attribute__((ext_vector_type(16))) __bf16 v16bf;
typedef __attribute__((ext_vector_type(8)))  float  v8f;
typedef __attribute__((ext_vector_type(8)))  unsigned short ushort8;

union FragU {
  v16bf   v;
  ushort8 h[2];
};

__device__ __forceinline__ unsigned short f2bf(float f) {
  unsigned int u = __float_as_uint(f);
  u += 0x7FFFu + ((u >> 16) & 1u);   // round-to-nearest-even
  return (unsigned short)(u >> 16);
}

#define B_BATCH 4
#define T_SEQ   2048
#define D_MODEL 1024
#define KD      512
#define OUT_D   1536   // D_MODEL + KD (concat output)

// ---------------------------------------------------------------------------
// Kernel 1: inputs -> bf16 workspace, and fp32 passthrough into out[:, :, 0:1024]
// ---------------------------------------------------------------------------
__global__ __launch_bounds__(256) void prep_inputs_kernel(
    const float* __restrict__ inp, unsigned short* __restrict__ in_bf,
    float* __restrict__ out) {
  int i = blockIdx.x * 256 + threadIdx.x;      // exact grid: B*T*D / 256
  float x = inp[i];
  in_bf[i] = f2bf(x);
  int row = i >> 10;              // / D_MODEL
  int col = i & (D_MODEL - 1);
  out[(size_t)row * OUT_D + col] = x;
}

// ---------------------------------------------------------------------------
// Kernel 2: Wq/Wk/Wv [D, 512] f32 -> transposed bf16 Wt[mat][n][d]
// ---------------------------------------------------------------------------
__global__ __launch_bounds__(256) void prep_weights_kernel(
    const float* __restrict__ Wq, const float* __restrict__ Wk,
    const float* __restrict__ Wv, unsigned short* __restrict__ Wt) {
  int i = blockIdx.x * 256 + threadIdx.x;      // exact grid: 3*D*KD / 256
  int mat = i / (D_MODEL * KD);
  int rem = i - mat * (D_MODEL * KD);
  const float* W = (mat == 0) ? Wq : (mat == 1) ? Wk : Wv;
  int d = rem >> 9;               // / KD
  int n = rem & (KD - 1);
  Wt[(size_t)mat * (KD * D_MODEL) + (size_t)n * D_MODEL + d] = f2bf(W[rem]);
}

// ---------------------------------------------------------------------------
// Kernel 3: QKV projection GEMM via v_wmma_f32_16x16x32_bf16.
// One wave computes one 16x16 output tile, K = 1024 -> 32 WMMAs.
// Q,K stored row-major bf16 [B*T][512]; V stored transposed bf16 [B][512][T].
// ---------------------------------------------------------------------------
__global__ __launch_bounds__(256) void qkv_proj_kernel(
    const unsigned short* __restrict__ in_bf,
    const unsigned short* __restrict__ Wt,
    const float* __restrict__ bq, const float* __restrict__ bk,
    const float* __restrict__ bv,
    unsigned short* __restrict__ qws, unsigned short* __restrict__ kws,
    unsigned short* __restrict__ vtw) {
  const int lane = threadIdx.x & 31;
  const int wv   = threadIdx.x >> 5;
  const int gid  = blockIdx.x * 8 + wv;
  const int ROWT = (B_BATCH * T_SEQ) / 16;     // 512
  const int COLT = KD / 16;                    // 32
  int mat = gid / (ROWT * COLT);
  int rem = gid - mat * (ROWT * COLT);
  int rt  = rem / COLT;
  int ct  = rem - rt * COLT;
  const int m0 = rt * 16, n0 = ct * 16;
  const int l15 = lane & 15, half = lane >> 4;

  const float* bias = (mat == 0) ? bq : (mat == 1) ? bk : bv;
  const float bval  = bias[n0 + l15];          // C layout: col = lane&15
  v8f acc;
#pragma unroll
  for (int r = 0; r < 8; ++r) acc[r] = bval;   // bias folded into accumulator

  const unsigned short* arow = in_bf + (size_t)(m0 + l15) * D_MODEL;
  const unsigned short* brow =
      Wt + (size_t)mat * (KD * D_MODEL) + (size_t)(n0 + l15) * D_MODEL;

#pragma unroll 4
  for (int c = 0; c < D_MODEL / 32; ++c) {
    FragU A, Bf;
    const int ka = c * 32 + half * 8;          // A: K = e + half*8 (+16 for e>=8)
    A.h[0] = *(const ushort8*)(arow + ka);
    A.h[1] = *(const ushort8*)(arow + ka + 16);
    const int kb = c * 32 + half * 8;          // B: K = e + half*8 (+16 for e>=8)
    Bf.h[0] = *(const ushort8*)(brow + kb);
    Bf.h[1] = *(const ushort8*)(brow + kb + 16);
    acc = __builtin_amdgcn_wmma_f32_16x16x32_bf16(false, A.v, false, Bf.v,
                                                  (short)0, acc, false, false);
  }

  if (mat == 2) {   // V: store transposed vt[b][n][t]
#pragma unroll
    for (int r = 0; r < 8; ++r) {
      int row = m0 + r + half * 8;             // C layout: row = r + half*8
      int b = row >> 11, t = row & (T_SEQ - 1);
      vtw[(size_t)(b * KD + n0 + l15) * T_SEQ + t] = f2bf(acc[r]);
    }
  } else {
    unsigned short* dst = (mat == 0) ? qws : kws;
#pragma unroll
    for (int r = 0; r < 8; ++r) {
      int row = m0 + r + half * 8;
      dst[(size_t)row * KD + n0 + l15] = f2bf(acc[r]);
    }
  }
}

// ---------------------------------------------------------------------------
// Kernel 4: causal flash attention. Block = 64 threads (2 waves), one 16-query
// tile per block; each wave owns 256 of the 512 V-columns (16 C-tiles).
// Key loop in tiles of 32 with online softmax; P transposed via LDS.
// ---------------------------------------------------------------------------
__global__ __launch_bounds__(64) void attn_kernel(
    const unsigned short* __restrict__ qws,
    const unsigned short* __restrict__ kws,
    const unsigned short* __restrict__ vtw,
    float* __restrict__ out) {
  __shared__ __align__(16) unsigned short sQ[16 * KD];       // 16 KB
  __shared__ __align__(16) unsigned short sP[2][16 * 32];    // per-wave P tile

  const int lane = threadIdx.x & 31;
  const int wv   = threadIdx.x >> 5;
  const int l15  = lane & 15, half = lane >> 4;

  const int bt = blockIdx.x;
  const int b  = bt >> 7;          // 128 query tiles per batch
  const int qt = bt & 127;
  const int t0 = qt * 16;

  // cooperative copy: 16 query rows are contiguous in qws -> flat 16 KB copy
  {
    const ushort8* src = (const ushort8*)(qws + (size_t)(b * T_SEQ + t0) * KD);
    ushort8* dst = (ushort8*)sQ;
    for (int i = threadIdx.x; i < (16 * KD) / 8; i += 64) dst[i] = src[i];
  }
  __syncthreads();

  const int n0w = wv * 256;
  v8f O[16];
#pragma unroll
  for (int nt = 0; nt < 16; ++nt)
#pragma unroll
    for (int r = 0; r < 8; ++r) O[nt][r] = 0.0f;

  float mrow[8], lrow[8];
#pragma unroll
  for (int r = 0; r < 8; ++r) { mrow[r] = -3.4e38f; lrow[r] = 0.0f; }

  const float scale = 0.0441941738241592f;     // 1/sqrt(512)
  const int nkt = (t0 + 15) / 32 + 1;          // causal key-tile count

  for (int kt = 0; kt < nkt; ++kt) {
    const int s0 = kt * 32;
    v8f S0, S1;
#pragma unroll
    for (int r = 0; r < 8; ++r) { S0[r] = 0.0f; S1[r] = 0.0f; }

    const unsigned short* arow = sQ + l15 * KD;
    const unsigned short* kr0 = kws + (size_t)(b * T_SEQ + s0 + l15) * KD;
    const unsigned short* kr1 = kws + (size_t)(b * T_SEQ + s0 + 16 + l15) * KD;
    if (kt + 1 < nkt)  // warm L2/WGP$ for the next key tile
      __builtin_prefetch(kws + (size_t)(b * T_SEQ + s0 + 32 + l15) * KD, 0, 1);

#pragma unroll 4
    for (int c = 0; c < KD / 32; ++c) {        // S = Q * K^T (32 WMMAs/tile)
      FragU A, B0, B1;
      const int ka = c * 32 + half * 8;
      A.h[0] = *(const ushort8*)(arow + ka);
      A.h[1] = *(const ushort8*)(arow + ka + 16);
      const int kb = c * 32 + half * 8;
      B0.h[0] = *(const ushort8*)(kr0 + kb);
      B0.h[1] = *(const ushort8*)(kr0 + kb + 16);
      B1.h[0] = *(const ushort8*)(kr1 + kb);
      B1.h[1] = *(const ushort8*)(kr1 + kb + 16);
      S0 = __builtin_amdgcn_wmma_f32_16x16x32_bf16(false, A.v, false, B0.v,
                                                   (short)0, S0, false, false);
      S1 = __builtin_amdgcn_wmma_f32_16x16x32_bf16(false, A.v, false, B1.v,
                                                   (short)0, S1, false, false);
    }

    // causal mask + scale + online softmax (row = r + half*8 across 16 lanes)
#pragma unroll
    for (int r = 0; r < 8; ++r) {
      const int row = t0 + r + half * 8;
      float x0 = ((s0 + l15)      <= row) ? S0[r] * scale : -3.4e38f;
      float x1 = ((s0 + 16 + l15) <= row) ? S1[r] * scale : -3.4e38f;
      float tm = fmaxf(x0, x1);
      tm = fmaxf(tm, __shfl_xor(tm, 1, 32));
      tm = fmaxf(tm, __shfl_xor(tm, 2, 32));
      tm = fmaxf(tm, __shfl_xor(tm, 4, 32));
      tm = fmaxf(tm, __shfl_xor(tm, 8, 32));
      const float mnew  = fmaxf(mrow[r], tm);
      const float alpha = __expf(mrow[r] - mnew);
      const float p0 = __expf(x0 - mnew);
      const float p1 = __expf(x1 - mnew);
      float ps = p0 + p1;
      ps += __shfl_xor(ps, 1, 32);
      ps += __shfl_xor(ps, 2, 32);
      ps += __shfl_xor(ps, 4, 32);
      ps += __shfl_xor(ps, 8, 32);
      lrow[r] = lrow[r] * alpha + ps;
      mrow[r] = mnew;
#pragma unroll
      for (int nt = 0; nt < 16; ++nt) O[nt][r] *= alpha;   // rescale O rows
      const int rl = r + half * 8;
      sP[wv][rl * 32 + l15]      = f2bf(p0);   // C layout -> LDS (row-major)
      sP[wv][rl * 32 + 16 + l15] = f2bf(p1);
    }
    __syncthreads();   // order P LDS writes before A-layout reads

    FragU PA;          // P as A-fragment: lane m = l15, keys e + half*8 (+16)
    const unsigned short* prow = &sP[wv][l15 * 32];
    PA.h[0] = *(const ushort8*)(prow + half * 8);
    PA.h[1] = *(const ushort8*)(prow + half * 8 + 16);

#pragma unroll
    for (int nt = 0; nt < 16; ++nt) {          // O += P * V (16 WMMAs/tile)
      const int n = n0w + nt * 16 + l15;
      const unsigned short* vrow = vtw + (size_t)(b * KD + n) * T_SEQ + s0;
      FragU VB;
      VB.h[0] = *(const ushort8*)(vrow + half * 8);
      VB.h[1] = *(const ushort8*)(vrow + half * 8 + 16);
      O[nt] = __builtin_amdgcn_wmma_f32_16x16x32_bf16(false, PA.v, false, VB.v,
                                                      (short)0, O[nt], false, false);
    }
  }

  // normalize and write out[:, :, 1024:1536]
#pragma unroll
  for (int nt = 0; nt < 16; ++nt) {
    const int n = n0w + nt * 16 + l15;
#pragma unroll
    for (int r = 0; r < 8; ++r) {
      const int t = t0 + r + half * 8;
      out[(size_t)(b * T_SEQ + t) * OUT_D + D_MODEL + n] = O[nt][r] / lrow[r];
    }
  }
}

// ---------------------------------------------------------------------------
extern "C" void kernel_launch(void* const* d_in, const int* in_sizes, int n_in,
                              void* d_out, int out_size, void* d_ws,
                              size_t ws_size, hipStream_t stream) {
  const float* inputs = (const float*)d_in[0];
  const float* Wq = (const float*)d_in[1];
  const float* bq = (const float*)d_in[2];
  const float* Wk = (const float*)d_in[3];
  const float* bk = (const float*)d_in[4];
  const float* Wv = (const float*)d_in[5];
  const float* bv = (const float*)d_in[6];
  float* out = (float*)d_out;

  char* ws = (char*)d_ws;
  unsigned short* in_bf = (unsigned short*)(ws);                            // 16 MB
  unsigned short* qws   = (unsigned short*)(ws + (size_t)16 * 1024 * 1024); //  8 MB
  unsigned short* kws   = (unsigned short*)(ws + (size_t)24 * 1024 * 1024); //  8 MB
  unsigned short* vtw   = (unsigned short*)(ws + (size_t)32 * 1024 * 1024); //  8 MB
  unsigned short* Wt    = (unsigned short*)(ws + (size_t)40 * 1024 * 1024); //  3 MB

  prep_inputs_kernel<<<(B_BATCH * T_SEQ * D_MODEL) / 256, 256, 0, stream>>>(
      inputs, in_bf, out);
  prep_weights_kernel<<<(3 * D_MODEL * KD) / 256, 256, 0, stream>>>(
      Wq, Wk, Wv, Wt);
  // 3 mats * 512 row-tiles * 32 col-tiles = 49152 wave tasks, 8 waves/block
  qkv_proj_kernel<<<49152 / 8, 256, 0, stream>>>(in_bf, Wt, bq, bk, bv,
                                                 qws, kws, vtw);
  attn_kernel<<<(B_BATCH * T_SEQ) / 16, 64, 0, stream>>>(qws, kws, vtw, out);
}
